// NeuralCDE_25366076850737
// MI455X (gfx1250) — matrix operand order repaired and checked by reading, not compiled, not run on hardware
//
#include <hip/hip_runtime.h>
#include <hip/hip_bf16.h>

// CDNA5 / gfx1250: wave32, WMMA 16x16x32 bf16 -> f32 accum, async global->LDS.
typedef __attribute__((ext_vector_type(16))) __bf16 v16bf;
typedef __attribute__((ext_vector_type(8)))  __bf16 v8bf;
typedef __attribute__((ext_vector_type(8)))  float  v8f;
typedef int v4i __attribute__((vector_size(16)));   // matches builtin param type

#define AS1 __attribute__((address_space(1)))
#define AS3 __attribute__((address_space(3)))

#if defined(__has_builtin)
#if __has_builtin(__builtin_amdgcn_global_load_async_to_lds_b128) && \
    __has_builtin(__builtin_amdgcn_s_wait_asynccnt)
#define USE_ASYNC_LDS 1
#endif
#endif

#define B_   256
#define L_   128
#define IN_  32
#define H_   128
#define HH_  128
#define OUT_ 32
#define MT_  16          // batch-tile rows per workgroup (WMMA M)

// ---------------- LDS layout (dynamic shared) ----------------
// bf16 wlds [3][8 nt][4 kt][512]   =  98304 B  (W1..W3 packed fragments)
// bf16 xbuf [16][128]              =   4096 B  (current layer input, bf16)
// f32  hst/hnx/kbf [16][128] x3    =  24576 B  (h state, RK4 accum, k stage)
// f32  da/de [16][32] x2           =   4096 B  (interval slopes)
// f32  ytl  [16][256]              =  16384 B  (tanh chunk for einsum)
// bf16 w4buf[2][16 nt][4 kt][512]  = 131072 B  (async double buffer for W4)
#define SMEM_W123   98304
#define XBUF_OFF    SMEM_W123                      // bytes
#define F32_OFF     (SMEM_W123 + 4096)             // bytes
#define W4BUF_OFF   147456                         // bytes
#define SMEM_BYTES  278528
// f32 region offsets (in floats, from F32_OFF)
#define HST_F   0
#define HNX_F   2048
#define KBF_F   4096
#define DA_F    6144
#define DE_F    6656
#define YTL_F   7168
#define W4_CHUNK_ELEMS 32768   // 16 nt * 4 kt * 512 bf16 = 64 KB per chunk

// ---- prepack: row-major f32 W(K,N) -> bf16 WMMA B-fragments ----
// fragment index = nt*(K/32)+kt ; element (k,n): lane=(n&15)|(((k>>4)&1)<<4), slot=k&15
__global__ void pack_b_kernel(const float* __restrict__ W, __bf16* __restrict__ dst,
                              int K, int N) {
    int idx = blockIdx.x * blockDim.x + threadIdx.x;
    if (idx >= K * N) return;
    int k = idx / N, n = idx % N;
    int kt = k >> 5, nt = n >> 4;
    int lane = (n & 15) | (((k >> 4) & 1) << 4);
    int slot = k & 15;
    int frag = nt * (K >> 5) + kt;
    dst[(frag << 9) + lane * 16 + slot] = (__bf16)W[idx];
}

// A fragment (16x32 bf16) from row-major LDS tile (stride 128):
// lane m=lane&15, hi=lane>>4: slots 0-15 = k[hi*16..+16)
__device__ __forceinline__ v16bf load_afrag(const __bf16* x, int kt, int lane) {
    int m = lane & 15, hi = lane >> 4;
    const __bf16* base = x + m * 128 + kt * 32 + hi * 16;
    v8bf lo = *(const v8bf*)(base);
    v8bf hh = *(const v8bf*)(base + 8);
    return __builtin_shufflevector(lo, hh, 0,1,2,3,4,5,6,7,8,9,10,11,12,13,14,15);
}

__device__ __forceinline__ v8f wmma_bf16(v16bf a, v16bf b, v8f c) {
    return __builtin_amdgcn_wmma_f32_16x16x32_bf16(false, a, false, b, (short)0, c,
                                                   false, false);
}

#if defined(USE_ASYNC_LDS)
// Issue one 64 KB W4 chunk copy: 256 threads x 16 async b128 (global -> LDS).
__device__ __forceinline__ void issue_w4_chunk(const __bf16* W4_p, int chunk,
                                               __bf16* dstbuf, int tid) {
    const char* g = (const char*)(W4_p + (size_t)chunk * W4_CHUNK_ELEMS) + tid * 16;
    char* l = (char*)dstbuf + tid * 16;
    #pragma unroll
    for (int t = 0; t < 16; ++t)
        __builtin_amdgcn_global_load_async_to_lds_b128(
            (AS1 v4i*)(g + t * 4096), (AS3 v4i*)(l + t * 4096), 0, 0);
}
#endif

// One vector-field evaluation for this workgroup's 16-row batch tile.
// in:  xbuf = bf16(input h'), 16x128    out: kbf = einsum(tanh(mlp(h')), slope)
__device__ __attribute__((noinline))
void vf_eval(const __bf16* __restrict__ W4_p,
             const float* __restrict__ b1, const float* __restrict__ b2,
             const float* __restrict__ b3, const float* __restrict__ b4,
             int use_de, int lane, int wave, int tid) {
    extern __shared__ char smem[];
    __bf16* wlds  = (__bf16*)smem;
    __bf16* xbuf  = (__bf16*)(smem + XBUF_OFF);
    float*  fb    = (float*)(smem + F32_OFF);
    float*  kbf   = fb + KBF_F;
    float*  ytl   = fb + YTL_F;
    const float* slope = fb + (use_de ? DE_F : DA_F);
    __bf16* w4buf = (__bf16*)(smem + W4BUF_OFF);
    const int nl = lane & 15, hi = lane >> 4;

    // ---- three hidden Linear+ReLU layers (M=16,K=128,N=128; 1 n-tile/wave) ----
    for (int l = 0; l < 3; ++l) {
        v16bf a0 = load_afrag(xbuf, 0, lane);
        v16bf a1 = load_afrag(xbuf, 1, lane);
        v16bf a2 = load_afrag(xbuf, 2, lane);
        v16bf a3 = load_afrag(xbuf, 3, lane);
        __syncthreads();                       // all reads done before overwrite
        const float* bias = (l == 0) ? b1 : ((l == 1) ? b2 : b3);
        const __bf16* wl = wlds + l * 16384 + wave * (4 * 512);
        v8f c = {};
        c = wmma_bf16(a0, *(const v16bf*)(wl + 0 * 512 + lane * 16), c);
        c = wmma_bf16(a1, *(const v16bf*)(wl + 1 * 512 + lane * 16), c);
        c = wmma_bf16(a2, *(const v16bf*)(wl + 2 * 512 + lane * 16), c);
        c = wmma_bf16(a3, *(const v16bf*)(wl + 3 * 512 + lane * 16), c);
        float bv = bias[wave * 16 + nl];
        #pragma unroll
        for (int j = 0; j < 8; ++j) {
            float v = c[j] + bv;
            v = v > 0.f ? v : 0.f;             // ReLU
            xbuf[(j + 8 * hi) * 128 + wave * 16 + nl] = (__bf16)v;
        }
        __syncthreads();
    }

    // ---- output layer W4 (M=16,K=128,N=4096), fused tanh + batched einsum ----
    v16bf a0 = load_afrag(xbuf, 0, lane);
    v16bf a1 = load_afrag(xbuf, 1, lane);
    v16bf a2 = load_afrag(xbuf, 2, lane);
    v16bf a3 = load_afrag(xbuf, 3, lane);
#if defined(USE_ASYNC_LDS)
    issue_w4_chunk(W4_p, 0, w4buf, tid);       // prefetch chunk 0 into buf 0
#endif
    #pragma unroll 1
    for (int c4 = 0; c4 < 16; ++c4) {          // 16 chunks of 256 columns (8 h each)
#if defined(USE_ASYNC_LDS)
        if (c4 < 15) {                          // prefetch next chunk, wait current
            issue_w4_chunk(W4_p, c4 + 1, w4buf + ((c4 + 1) & 1) * W4_CHUNK_ELEMS, tid);
            __builtin_amdgcn_s_wait_asynccnt(16);
        } else {
            __builtin_amdgcn_s_wait_asynccnt(0);
        }
        __syncthreads();                        // whole chunk visible to all waves
        const __bf16* wcb = w4buf + (c4 & 1) * W4_CHUNK_ELEMS;
#endif
        #pragma unroll
        for (int t = 0; t < 2; ++t) {          // 2 n-tiles per wave per chunk
            int ntl = wave * 2 + t;            // n-tile within chunk
            int ntg = c4 * 16 + ntl;           // global n-tile
#if defined(USE_ASYNC_LDS)
            const __bf16* wp = wcb + ntl * (4 * 512);        // LDS (ds_load_b128)
#else
            const __bf16* wp = W4_p + (size_t)ntg * (4 * 512); // L2 stream fallback
#endif
            v8f acc = {};
            acc = wmma_bf16(a0, *(const v16bf*)(wp + 0 * 512 + lane * 16), acc);
            acc = wmma_bf16(a1, *(const v16bf*)(wp + 1 * 512 + lane * 16), acc);
            acc = wmma_bf16(a2, *(const v16bf*)(wp + 2 * 512 + lane * 16), acc);
            acc = wmma_bf16(a3, *(const v16bf*)(wp + 3 * 512 + lane * 16), acc);
            float bv = b4[ntg * 16 + nl];
            int ncol = wave * 32 + t * 16 + nl;
            #pragma unroll
            for (int j = 0; j < 8; ++j)
                ytl[(j + 8 * hi) * 256 + ncol] = tanhf(acc[j] + bv);
        }
        __syncthreads();
        // einsum: k[m, h] = sum_i y[m, h, i] * slope[m, i]  (8 h values per chunk)
        if (tid < 128) {
            int m = tid >> 3, hh = tid & 7;
            const float* yr = ytl + m * 256 + hh * 32;
            const float* sr = slope + m * 32;
            float s = 0.f;
            #pragma unroll
            for (int i = 0; i < 32; ++i) s += yr[i] * sr[i];
            kbf[m * 128 + c4 * 8 + hh] = s;
        }
        __syncthreads();                        // ytl / w4buf[c4&1] free for reuse
    }
}

__launch_bounds__(256, 1)
__global__ void cde_kernel(const float* __restrict__ initial,
                           const float* __restrict__ coeffs,
                           const __bf16* __restrict__ Wi_p, const float* __restrict__ bi,
                           const __bf16* __restrict__ W123_p,
                           const float* __restrict__ b1, const float* __restrict__ b2,
                           const float* __restrict__ b3,
                           const __bf16* __restrict__ W4_p, const float* __restrict__ b4,
                           const __bf16* __restrict__ Wf_p, const float* __restrict__ bfin,
                           float* __restrict__ out) {
    extern __shared__ char smem[];
    __bf16* wlds = (__bf16*)smem;
    __bf16* xbuf = (__bf16*)(smem + XBUF_OFF);
    float*  fb   = (float*)(smem + F32_OFF);
    float*  hst  = fb + HST_F;
    float*  hnx  = fb + HNX_F;
    float*  kbf  = fb + KBF_F;
    float*  da   = fb + DA_F;
    float*  de   = fb + DE_F;

    const int tid = threadIdx.x, lane = tid & 31, wave = tid >> 5;
    const int nl = lane & 15, hi = lane >> 4;
    const int b0 = blockIdx.x * MT_;

    // Stage packed W1..W3 fragments into LDS once (L2 -> LDS, 96KB).
    {
        const uint4* src = (const uint4*)W123_p;
        uint4* dst = (uint4*)wlds;
        for (int i = tid; i < SMEM_W123 / 16; i += 256) dst[i] = src[i];
    }
    // initial tile -> xbuf (bf16), cols 0..31
    for (int e = tid; e < MT_ * IN_; e += 256) {
        int m = e >> 5, i = e & 31;
        xbuf[m * 128 + i] = (__bf16)initial[(b0 + m) * IN_ + i];
    }
    __syncthreads();

    // h0 = initial @ Wi + bi   (K=32 -> single k-tile, one n-tile per wave)
    {
        v16bf a = load_afrag(xbuf, 0, lane);
        v8f c = {};
        c = wmma_bf16(a, *(const v16bf*)(Wi_p + (wave << 9) + lane * 16), c);
        float bv = bi[wave * 16 + nl];
        #pragma unroll
        for (int j = 0; j < 8; ++j)
            hst[(j + 8 * hi) * 128 + wave * 16 + nl] = c[j] + bv;
    }
    __syncthreads();

    // ---------------- RK4 over the 127 intervals ----------------
    for (int step = 0; step < L_ - 1; ++step) {
        for (int e = tid; e < MT_ * IN_ * 2; e += 256) {
            int ee = e & 511;
            int m = ee >> 5, i = ee & 31;
            const float* cb = coeffs + (size_t)(b0 + m) * L_ * IN_;
            if (e < 512) {
                da[ee] = cb[(step + 1) * IN_ + i] - cb[step * IN_ + i];
            } else {
                int s2 = (step + 1 < L_ - 1) ? step + 1 : L_ - 2;
                de[ee] = cb[(s2 + 1) * IN_ + i] - cb[s2 * IN_ + i];
            }
        }
        for (int e = tid; e < 2048; e += 256) xbuf[e] = (__bf16)hst[e];
        __syncthreads();
        vf_eval(W4_p, b1, b2, b3, b4, 0, lane, wave, tid);               // k1
        for (int e = tid; e < 2048; e += 256) {
            float h = hst[e], k = kbf[e];
            hnx[e] = h + k * (1.f / 6.f);
            xbuf[e] = (__bf16)(h + 0.5f * k);
        }
        __syncthreads();
        vf_eval(W4_p, b1, b2, b3, b4, 0, lane, wave, tid);               // k2
        for (int e = tid; e < 2048; e += 256) {
            float k = kbf[e];
            hnx[e] += k * (1.f / 3.f);
            xbuf[e] = (__bf16)(hst[e] + 0.5f * k);
        }
        __syncthreads();
        vf_eval(W4_p, b1, b2, b3, b4, 0, lane, wave, tid);               // k3
        for (int e = tid; e < 2048; e += 256) {
            float k = kbf[e];
            hnx[e] += k * (1.f / 3.f);
            xbuf[e] = (__bf16)(hst[e] + k);
        }
        __syncthreads();
        vf_eval(W4_p, b1, b2, b3, b4, 1, lane, wave, tid);               // k4 (de)
        for (int e = tid; e < 2048; e += 256)
            hst[e] = hnx[e] + kbf[e] * (1.f / 6.f);
        __syncthreads();
    }

    // ---------------- final: out = hT @ Wf + bf ----------------
    for (int e = tid; e < 2048; e += 256) xbuf[e] = (__bf16)hst[e];
    __syncthreads();
    if (wave < 2) {                               // OUT=32 -> 2 n-tiles
        v16bf a0 = load_afrag(xbuf, 0, lane);
        v16bf a1 = load_afrag(xbuf, 1, lane);
        v16bf a2 = load_afrag(xbuf, 2, lane);
        v16bf a3 = load_afrag(xbuf, 3, lane);
        const __bf16* wp = Wf_p + wave * (4 * 512);
        v8f c = {};
        c = wmma_bf16(a0, *(const v16bf*)(wp + 0 * 512 + lane * 16), c);
        c = wmma_bf16(a1, *(const v16bf*)(wp + 1 * 512 + lane * 16), c);
        c = wmma_bf16(a2, *(const v16bf*)(wp + 2 * 512 + lane * 16), c);
        c = wmma_bf16(a3, *(const v16bf*)(wp + 3 * 512 + lane * 16), c);
        float bv = bfin[wave * 16 + nl];
        #pragma unroll
        for (int j = 0; j < 8; ++j)
            out[(size_t)(b0 + j + 8 * hi) * OUT_ + wave * 16 + nl] = c[j] + bv;
    }
}

extern "C" void kernel_launch(void* const* d_in, const int* in_sizes, int n_in,
                              void* d_out, int out_size, void* d_ws, size_t ws_size,
                              hipStream_t stream) {
    (void)in_sizes; (void)n_in; (void)out_size; (void)ws_size;
    const float* initial = (const float*)d_in[0];
    const float* coeffs  = (const float*)d_in[1];
    const float* Wi = (const float*)d_in[2];  const float* bi = (const float*)d_in[3];
    const float* W1 = (const float*)d_in[4];  const float* b1 = (const float*)d_in[5];
    const float* W2 = (const float*)d_in[6];  const float* b2 = (const float*)d_in[7];
    const float* W3 = (const float*)d_in[8];  const float* b3 = (const float*)d_in[9];
    const float* W4 = (const float*)d_in[10]; const float* b4 = (const float*)d_in[11];
    const float* Wf = (const float*)d_in[12]; const float* bf = (const float*)d_in[13];
    float* out = (float*)d_out;

    // workspace carve (bf16 packed fragments), ~1.11 MB total
    char* ws = (char*)d_ws;
    __bf16* Wi_p   = (__bf16*)(ws);                               // 8 KB
    __bf16* W123_p = (__bf16*)(ws + 8192);                        // 96 KB
    __bf16* W4_p   = (__bf16*)(ws + 8192 + 98304);                // 1 MB
    __bf16* Wf_p   = (__bf16*)(ws + 8192 + 98304 + 1048576);      // 8 KB

    pack_b_kernel<<<(32 * 128 + 255) / 256, 256, 0, stream>>>(Wi, Wi_p, 32, 128);
    pack_b_kernel<<<(128 * 128 + 255) / 256, 256, 0, stream>>>(W1, W123_p + 0 * 16384, 128, 128);
    pack_b_kernel<<<(128 * 128 + 255) / 256, 256, 0, stream>>>(W2, W123_p + 1 * 16384, 128, 128);
    pack_b_kernel<<<(128 * 128 + 255) / 256, 256, 0, stream>>>(W3, W123_p + 2 * 16384, 128, 128);
    pack_b_kernel<<<(128 * 4096 + 255) / 256, 256, 0, stream>>>(W4, W4_p, 128, 4096);
    pack_b_kernel<<<(128 * 32 + 255) / 256, 256, 0, stream>>>(Wf, Wf_p, 128, 32);

    static bool attr_set = [] {
        (void)hipFuncSetAttribute((const void*)cde_kernel,
                                  hipFuncAttributeMaxDynamicSharedMemorySize, SMEM_BYTES);
        return true;
    }();
    (void)attr_set;

    cde_kernel<<<B_ / MT_, 256, SMEM_BYTES, stream>>>(
        initial, coeffs, Wi_p, bi, W123_p, b1, b2, b3, W4_p, b4, Wf_p, bf, out);
}
